// Attention_62294205661435
// MI455X (gfx1250) — matrix operand order repaired and checked by reading, not compiled, not run on hardware
//
#include <hip/hip_runtime.h>
#include <cstdint>

// ---------------------------------------------------------------------------
// Fused segment-attention for MI455X (gfx1250, wave32, WMMA bf16).
//   logits = leaky(leaky(leaky([a,q,a-q,a*q] @ w0) @ w1) @ w2),  a = kv @ Wk
//   out    = kv * segment_softmax(logits)
// ---------------------------------------------------------------------------

#define NEG_SLOPE 0.01f

typedef __bf16 bf16_t;
typedef bf16_t v16bf __attribute__((ext_vector_type(16)));
typedef float  v8f   __attribute__((ext_vector_type(8)));

#if defined(__HIP_DEVICE_COMPILE__) && __has_builtin(__builtin_amdgcn_tensor_load_to_lds) && __has_builtin(__builtin_amdgcn_s_wait_tensorcnt)
#define HAVE_TDM 1
#else
#define HAVE_TDM 0
#endif

// ---------------- scalar helpers ----------------
__device__ __forceinline__ unsigned short f2bf(float f) {
  unsigned u = __float_as_uint(f);
  u += 0x7FFFu + ((u >> 16) & 1u);          // round-to-nearest-even
  return (unsigned short)(u >> 16);
}
__device__ __forceinline__ float bf2f(unsigned short h) {
  return __uint_as_float(((unsigned)h) << 16);
}
__device__ __forceinline__ float leaky(float x) { return x >= 0.f ? x : NEG_SLOPE * x; }

__device__ __forceinline__ v8f zero8() {
  v8f z;
  for (int i = 0; i < 8; ++i) z[i] = 0.f;
  return z;
}

// ---------------- WMMA fragment loaders (bf16, 16x16x32) ----------------
// A (16xK slice, row-major in LDS): lane m=l&15, hi=l>>4 holds
//   K = kbase+hi*16 .. +15  -> two 16B ds loads.
// B (KxN, stored N-major [n][k] in LDS): lane n=l&15, hi=l>>4 holds
//   K = kbase+hi*16 .. +15 of column n0+n      -> one 32B contiguous read.
union FragCast { uint4 q[2]; v16bf v; };

__device__ __forceinline__ v16bf frag_a16(const unsigned short* base, int rowStride, int kbase) {
  const int lane = threadIdx.x & 31;
  const int m = lane & 15, hi = lane >> 4;
  const unsigned short* p = base + m * rowStride + kbase + hi * 16;
  FragCast f;
  f.q[0] = *(const uint4*)(p);
  f.q[1] = *(const uint4*)(p + 8);
  return f.v;
}
__device__ __forceinline__ v16bf frag_b16(const unsigned short* baseT, int rowStride, int n0, int kbase) {
  const int lane = threadIdx.x & 31;
  const int n = lane & 15, hi = lane >> 4;
  const unsigned short* p = baseT + (n0 + n) * rowStride + kbase + hi * 16;
  FragCast f;
  f.q[0] = *(const uint4*)(p);
  f.q[1] = *(const uint4*)(p + 8);
  return f.v;
}
__device__ __forceinline__ v8f wmma_bf16(v16bf a, v16bf b, v8f c) {
  return __builtin_amdgcn_wmma_f32_16x16x32_bf16(false, a, false, b, (short)0, c, false, false);
}

// ---------------- TDM (Tensor Data Mover) 2D tile -> LDS ----------------
#if HAVE_TDM
typedef unsigned int tdm_v4u __attribute__((ext_vector_type(4)));
typedef int          tdm_v8i __attribute__((ext_vector_type(8)));
typedef int          tdm_v4i __attribute__((ext_vector_type(4)));

// bf16 2D tile: tile0 elems/row, tile1 rows, global row stride stride0 elems.
// LDS rows padded via TDM pad: pad_amount+1 DWORDs every 2^(pad_interval+1) DWORDs.
__device__ __forceinline__ void tdm_load_2d_bf16(const void* lds_dst, const void* gsrc,
                                                 unsigned tile0, unsigned tile1,
                                                 unsigned stride0,
                                                 unsigned pad_interval, unsigned pad_amount) {
  unsigned long long ga = (unsigned long long)(uintptr_t)gsrc;
  unsigned lds = (unsigned)(uintptr_t)lds_dst;      // low 32 bits = LDS byte offset
  tdm_v4u g0;
  g0[0] = 1u;                                       // count=1, user descriptor
  g0[1] = lds;
  g0[2] = (unsigned)ga;
  g0[3] = ((unsigned)(ga >> 32) & 0x01FFFFFFu) | 0x80000000u;   // type=2 ("image")
  unsigned d0 = (1u << 16)                          // data_size = 2 bytes
              | (1u << 20)                          // pad_enable
              | (pad_interval << 22)
              | (pad_amount << 25);
  tdm_v8i g1;
  g1[0] = (int)d0;                                  // workgroup_mask=0 (not in cluster)
  g1[1] = (int)((tile0 & 0xFFFFu) << 16);           // tensor_dim0[15:0]
  g1[2] = (int)((tile0 >> 16) | ((tile1 & 0xFFFFu) << 16));  // tdim0 hi | tdim1 lo
  g1[3] = (int)((tile1 >> 16) | ((tile0 & 0xFFFFu) << 16));  // tdim1 hi | tile_dim0
  g1[4] = (int)(tile1 & 0xFFFFu);                   // tile_dim1 | tile_dim2=0
  g1[5] = (int)stride0;                             // tensor_dim0_stride[31:0]
  g1[6] = 0;                                        // stride0 hi | stride1 lo
  g1[7] = 0;
  tdm_v4i g2; g2[0] = 1; g2[1] = 0; g2[2] = 0; g2[3] = 0;   // tensor_dim2=1
  tdm_v4i g3; g3[0] = 0; g3[1] = 0; g3[2] = 0; g3[3] = 0;
  tdm_v8i gz;                                       // trailing group (zero-filled,
  for (int i = 0; i < 8; ++i) gz[i] = 0;            //  per probed 6-arg signature)
  __builtin_amdgcn_tensor_load_to_lds(g0, g1, g2, g3, gz, 0);
}
#endif

// Fallback cooperative copy with manual LDS row padding (blockDim.x == 128).
__device__ __forceinline__ void coop_copy_pad(unsigned short* dst, int dstStride,
                                              const unsigned short* src, int srcStride,
                                              int rows, int cols) {
  for (int e = (int)threadIdx.x * 8; e < rows * cols; e += 128 * 8) {
    int r = e / cols, c = e - r * cols;
    *(uint4*)(dst + r * dstStride + c) = *(const uint4*)(src + r * srcStride + c);
  }
}

// ---------------- kernel 1: weight convert + transpose to bf16 [N][K] ------
__global__ __launch_bounds__(256) void attn62_prep(
    const float* __restrict__ wkey, const float* __restrict__ w0,
    const float* __restrict__ w1,
    unsigned short* __restrict__ wkeyT, unsigned short* __restrict__ w0T,
    unsigned short* __restrict__ w1T)
{
  int t = blockIdx.x * 256 + threadIdx.x;
  if (t < 128 * 128) { int k = t >> 7, n = t & 127; wkeyT[n * 128 + k] = f2bf(wkey[t]); }
  if (t < 512 * 256) { int k = t >> 8, n = t & 255; w0T[n * 512 + k] = f2bf(w0[t]); }
  if (t < 256 * 128) { int k = t >> 7, n = t & 127; w1T[n * 256 + k] = f2bf(w1[t]); }
}

// ---------------- kernel 2: fused GEMM + MLP -> logits ---------------------
// 64 rows per block, 4 waves (wave32) x 16 rows each.
// Padded LDS row strides (elements): +8 bf16 (=16B = 4 DWORDs) per row.
#define S_WKEY 136   // 128+8
#define S_W1   264   // 256+8
#define S_W0C  136   // 128+8 (K-chunk of w0)
#define S_FE   520   // 512+8
#define S_H0   264   // 256+8
#define S_KV   136   // 128+8

__global__ __launch_bounds__(128) void attn62_fused(
    const float* __restrict__ kv, const float* __restrict__ query,
    const int* __restrict__ bix,
    const unsigned short* __restrict__ wkeyT_g,
    const unsigned short* __restrict__ w0T_g,
    const unsigned short* __restrict__ w1T_g,
    const float* __restrict__ w2_g,
    float* __restrict__ logits, int N)
{
  __shared__ unsigned short sWkeyT[128 * S_WKEY];  // 34.0 KB  Wk^T  [n][k]
  __shared__ unsigned short sW1T  [128 * S_W1];    // 66.0 KB  w1^T  [n][k]
  __shared__ unsigned short sW0c  [256 * S_W0C];   // 68.0 KB  w0^T chunk [n][k128]
  __shared__ unsigned short sFeats[64 * S_FE];     // 65.0 KB  feats / later h1
  __shared__ unsigned short sH0   [64 * S_H0];     // 33.0 KB  h0
  __shared__ unsigned short sKV   [64 * S_KV];     // 17.0 KB  kv tile (bf16)
  __shared__ float sW2[128];

  const int wave = threadIdx.x >> 5;
  const int lane = threadIdx.x & 31;
  const int m0 = wave * 16;                 // this wave's 16-row strip
  const int rowBase = blockIdx.x * 64;

  // ---- stage 0: persistent weights into LDS (TDM from wave 0) ----
#if HAVE_TDM
  if (wave == 0) {
    tdm_load_2d_bf16(sWkeyT, wkeyT_g, 128, 128, 128, /*pad 64DW*/5, /*4DW*/3);
    tdm_load_2d_bf16(sW1T,   w1T_g,   256, 128, 256, /*pad 128DW*/6, /*4DW*/3);
    __builtin_amdgcn_s_wait_tensorcnt(0);
  }
#else
  coop_copy_pad(sWkeyT, S_WKEY, wkeyT_g, 128, 128, 128);
  coop_copy_pad(sW1T,   S_W1,   w1T_g,   256, 128, 256);
#endif
  if (threadIdx.x < 128) sW2[threadIdx.x] = w2_g[threadIdx.x];

  // ---- per-wave: stage kv rows (bf16) and gathered q rows into LDS ----
  for (int i = 0; i < 16; ++i) {
    const int lrow = m0 + i;
    int g = rowBase + lrow; if (g > N - 1) g = N - 1;
    const float4 v = ((const float4*)kv)[(size_t)g * 32 + lane];
    ushort4 pk = make_ushort4(f2bf(v.x), f2bf(v.y), f2bf(v.z), f2bf(v.w));
    *(ushort4*)(sKV + lrow * S_KV + lane * 4) = pk;
    const int b = bix[g];
    const float4 qv = ((const float4*)query)[(size_t)b * 32 + lane];
    ushort4 qk = make_ushort4(f2bf(qv.x), f2bf(qv.y), f2bf(qv.z), f2bf(qv.w));
    *(ushort4*)(sFeats + lrow * S_FE + 128 + lane * 4) = qk;
  }
  __syncthreads();   // weights + tiles visible to all waves

  // ---- stage 1: a = kv16 @ Wkey   (16x128 @ 128x128), build feats ----
  v8f acc1[8];
#pragma unroll
  for (int nt = 0; nt < 8; ++nt) acc1[nt] = zero8();
#pragma unroll
  for (int kw = 0; kw < 4; ++kw) {
    v16bf a = frag_a16(sKV + m0 * S_KV, S_KV, kw * 32);
#pragma unroll
    for (int nt = 0; nt < 8; ++nt) {
      v16bf b = frag_b16(sWkeyT, S_WKEY, nt * 16, kw * 32);
      acc1[nt] = wmma_bf16(a, b, acc1[nt]);
    }
  }
  {
    const int n = lane & 15, hi = lane >> 4;
#pragma unroll
    for (int nt = 0; nt < 8; ++nt)
#pragma unroll
      for (int r = 0; r < 8; ++r) {
        const int row = m0 + r + hi * 8;
        const int col = nt * 16 + n;
        const float af = acc1[nt][r];
        const float qf = bf2f(sFeats[row * S_FE + 128 + col]);
        sFeats[row * S_FE +       col] = f2bf(af);        // a
        sFeats[row * S_FE + 256 + col] = f2bf(af - qf);   // a - q
        sFeats[row * S_FE + 384 + col] = f2bf(af * qf);   // a * q
      }
  }

  // ---- stage 2: h0 = leaky(feats @ w0)  (16x512 @ 512x256), K-chunked ----
  v8f accB[16];
#pragma unroll
  for (int nt = 0; nt < 16; ++nt) accB[nt] = zero8();
  for (int kk = 0; kk < 4; ++kk) {
    __syncthreads();                        // previous chunk consumed
#if HAVE_TDM
    if (wave == 0) {
      tdm_load_2d_bf16(sW0c, w0T_g + kk * 128, 128, 256, 512, 5, 3);
      __builtin_amdgcn_s_wait_tensorcnt(0);
    }
#else
    coop_copy_pad(sW0c, S_W0C, w0T_g + kk * 128, 512, 256, 128);
#endif
    __syncthreads();
#pragma unroll
    for (int kw = 0; kw < 4; ++kw) {
      v16bf a = frag_a16(sFeats + m0 * S_FE, S_FE, kk * 128 + kw * 32);
#pragma unroll
      for (int nt = 0; nt < 16; ++nt) {
        v16bf b = frag_b16(sW0c, S_W0C, nt * 16, kw * 32);
        accB[nt] = wmma_bf16(a, b, accB[nt]);
      }
    }
  }
  {
    const int n = lane & 15, hi = lane >> 4;
#pragma unroll
    for (int nt = 0; nt < 16; ++nt)
#pragma unroll
      for (int r = 0; r < 8; ++r)
        sH0[(m0 + r + hi * 8) * S_H0 + nt * 16 + n] = f2bf(leaky(accB[nt][r]));
  }

  // ---- stage 3: h1 = leaky(h0 @ w1)  (16x256 @ 256x128) ----
  v8f accC[8];
#pragma unroll
  for (int nt = 0; nt < 8; ++nt) accC[nt] = zero8();
#pragma unroll
  for (int kw = 0; kw < 8; ++kw) {
    v16bf a = frag_a16(sH0 + m0 * S_H0, S_H0, kw * 32);
#pragma unroll
    for (int nt = 0; nt < 8; ++nt) {
      v16bf b = frag_b16(sW1T, S_W1, nt * 16, kw * 32);
      accC[nt] = wmma_bf16(a, b, accC[nt]);
    }
  }
  {
    const int n = lane & 15, hi = lane >> 4;
#pragma unroll
    for (int nt = 0; nt < 8; ++nt)
#pragma unroll
      for (int r = 0; r < 8; ++r)
        sFeats[(m0 + r + hi * 8) * S_FE + nt * 16 + n] = f2bf(leaky(accC[nt][r]));  // h1
  }
  __syncthreads();

  // ---- stage 4: logits = leaky(h1 . w2) ----
  if (threadIdx.x < 64) {
    float s = 0.f;
#pragma unroll 8
    for (int k = 0; k < 128; ++k)
      s += bf2f(sFeats[threadIdx.x * S_FE + k]) * sW2[k];
    s = leaky(s);
    const int g = rowBase + (int)threadIdx.x;
    if (g < N) logits[g] = s;
  }
}

// ---------------- kernel 3: deterministic segment max/sum ------------------
__global__ __launch_bounds__(128) void attn62_seg_reduce(
    const float* __restrict__ logits, const int* __restrict__ bix,
    float* __restrict__ smax, float* __restrict__ ssum, int N)
{
  __shared__ float red[128];
  const int b = blockIdx.x;
  const int tid = threadIdx.x;
  int lo = 0, hi = N;                               // lower_bound(b)
  while (lo < hi) { int mid = (lo + hi) >> 1; if (bix[mid] < b) lo = mid + 1; else hi = mid; }
  const int start = lo;
  hi = N;                                           // lower_bound(b+1)
  while (lo < hi) { int mid = (lo + hi) >> 1; if (bix[mid] < b + 1) lo = mid + 1; else hi = mid; }
  const int end = lo;

  float m = -3.4e38f;
  for (int i = start + tid; i < end; i += 128) m = fmaxf(m, logits[i]);
  red[tid] = m; __syncthreads();
  for (int s = 64; s > 0; s >>= 1) { if (tid < s) red[tid] = fmaxf(red[tid], red[tid + s]); __syncthreads(); }
  const float M = red[0];
  __syncthreads();

  float acc = 0.f;
  for (int i = start + tid; i < end; i += 128) acc += __expf(logits[i] - M);
  red[tid] = acc; __syncthreads();
  for (int s = 64; s > 0; s >>= 1) { if (tid < s) red[tid] += red[tid + s]; __syncthreads(); }
  if (tid == 0) { smax[b] = M; ssum[b] = red[0]; }
}

// ---------------- kernel 4: scale kv rows by softmax weight ----------------
__global__ __launch_bounds__(256) void attn62_finalize(
    const float* __restrict__ kv, const int* __restrict__ bix,
    const float* __restrict__ logits, const float* __restrict__ smax,
    const float* __restrict__ ssum, float* __restrict__ out,
    float* __restrict__ attw, int N)
{
  const int row = blockIdx.x * 8 + ((int)threadIdx.x >> 5);
  if (row >= N) return;
  const int lane = threadIdx.x & 31;
  const int b = bix[row];
  const float w = __expf(logits[row] - smax[b]) / ssum[b];
  float4 v = ((const float4*)kv)[(size_t)row * 32 + lane];
  v.x *= w; v.y *= w; v.z *= w; v.w *= w;
  ((float4*)out)[(size_t)row * 32 + lane] = v;
  if (lane == 0) attw[row] = w;
}

// ---------------- launch ----------------
extern "C" void kernel_launch(void* const* d_in, const int* in_sizes, int n_in,
                              void* d_out, int out_size, void* d_ws, size_t ws_size,
                              hipStream_t stream) {
  (void)n_in; (void)out_size; (void)ws_size;
  const float* query = (const float*)d_in[0];   // [B,128]
  const float* kv    = (const float*)d_in[1];   // [N,128]
  const int*   bix   = (const int*)d_in[2];     // [N] sorted
  const float* wkey  = (const float*)d_in[3];   // [128,128]
  const float* w0    = (const float*)d_in[4];   // [512,256]
  const float* w1    = (const float*)d_in[5];   // [256,128]
  const float* w2    = (const float*)d_in[6];   // [128,1]
  const int N = in_sizes[2];
  const int B = in_sizes[0] / 128;

  char* ws = (char*)d_ws;                          // workspace layout (bytes)
  unsigned short* wkeyT = (unsigned short*)(ws + 0);        //  32768
  unsigned short* w0T   = (unsigned short*)(ws + 32768);    // 262144
  unsigned short* w1T   = (unsigned short*)(ws + 294912);   //  65536
  float* smax   = (float*)(ws + 360448);                    //  4*B (<=64KB)
  float* ssum   = (float*)(ws + 360448 + 65536);            //  4*B (<=64KB)
  float* logits = (float*)(ws + 491520);                    //  4*N

  float* out  = (float*)d_out;            // [N,128]
  float* attw = out + (size_t)N * 128;    // [N]

  attn62_prep<<<(512 * 256 + 255) / 256, 256, 0, stream>>>(wkey, w0, w1, wkeyT, w0T, w1T);
  attn62_fused<<<(N + 63) / 64, 128, 0, stream>>>(kv, query, bix, wkeyT, w0T, w1T, w2,
                                                  logits, N);
  attn62_seg_reduce<<<B, 128, 0, stream>>>(logits, bix, smax, ssum, N);
  attn62_finalize<<<(N + 7) / 8, 256, 0, stream>>>(kv, bix, logits, smax, ssum, out, attw, N);
}
